// Qwen3_5MLP_75917841924553
// MI455X (gfx1250) — matrix operand order repaired and checked by reading, not compiled, not run on hardware
//
#include <hip/hip_runtime.h>
#include <hip/hip_bf16.h>

#define HID 2048
#define INTER 6144

typedef __attribute__((ext_vector_type(16))) _Float16 v16h;
typedef __attribute__((ext_vector_type(8)))  _Float16 v8h;
typedef __attribute__((ext_vector_type(8)))  float    v8f;

union FragU { v16h v; v8h h[2]; };

// ---------------------------------------------------------------------------
// Kernel 1: LPBQ weight dequant: w[out_f, in_f] f32 + s[out_f, in_f/16] ->
// f16 fake-quant weight. One thread per 16-element block.
// ---------------------------------------------------------------------------
__global__ __launch_bounds__(256) void lpbq_dequant_kernel(
    const float* __restrict__ w, const float* __restrict__ s,
    _Float16* __restrict__ out, int in_f, long long nblocks) {
  long long b = blockIdx.x * 256LL + threadIdx.x;
  if (b >= nblocks) return;
  int bpr = in_f >> 4;
  long long row = b / bpr;
  int cb = (int)(b - row * bpr);
  float sc = s[row * bpr + cb];
  float inv = 1.0f / sc;
  const float* wp = w + row * (long long)in_f + cb * 16;
  _Float16* op = out + row * (long long)in_f + cb * 16;
#pragma unroll
  for (int i = 0; i < 16; ++i) {
    float q = rintf(wp[i] * inv);
    q = fminf(fmaxf(q, -8.0f), 7.0f);
    op[i] = (_Float16)(q * sc);
  }
}

// ---------------------------------------------------------------------------
// Kernel 2: 16-bit asymmetric activation QDQ of x -> f16.
// ---------------------------------------------------------------------------
__global__ __launch_bounds__(256) void act_qdq_kernel(
    const float* __restrict__ x, _Float16* __restrict__ out,
    const float* __restrict__ a_scales, long long n) {
  float sc = a_scales[0];
  float inv = 1.0f / sc;
  long long i = (blockIdx.x * 256LL + threadIdx.x) * 4;
  if (i >= n) return;
#pragma unroll
  for (int j = 0; j < 4; ++j) {
    float q = rintf(x[i + j] * inv) + 32768.0f;
    q = fminf(fmaxf(q, 0.0f), 65535.0f);
    out[i + j] = (_Float16)((q - 32768.0f) * sc);
  }
}

// ---------------------------------------------------------------------------
// Kernel 3/5: WMMA GEMM  C[M,N] = A[M,K] * B[N,K]^T  (A,B f16, acc f32)
// 128x256 C tile per 256-thread block. 8 waves in a 2(M) x 4(N) grid; each
// wave owns a 64x64 sub-block = 4x4 WMMA fragments (128 acc VGPRs).
// K stepped in 32 with LDS double-buffering + software pipeline:
//   store tile k -> buf; barrier; issue global loads for tile k+1 (drain under
//   loadcnt while computing); 16 x v_wmma from buf; flip.
// LDS tiles are stored in *fragment order* so each lane's v16h operand is a
// contiguous 32B run (two ds_load_b128):
//   A frag (16x32): lane = (m&15) + 16*((k>>3)&1), half = 8*(k>>4) + (k&7)
//   B frag (32x16): lane = (n&15) + 16*((k>>3)&1), half = 8*(k>>4) + (k&7)
// Both maps send an aligned 8-half global run to a contiguous 16B LDS run.
// ---------------------------------------------------------------------------
template <bool QDQ_OUT>
__global__ __launch_bounds__(256) void gemm_wmma_kernel(
    const _Float16* __restrict__ A, const _Float16* __restrict__ B,
    void* __restrict__ Cout, int M, int N, int K,
    const float* __restrict__ a_scales, int sidx) {
  // 24-half lane stride: 48B (16B-aligned, reduced bank conflicts)
  __shared__ __align__(16) _Float16 lds_a[2][8][32][24];
  __shared__ __align__(16) _Float16 lds_b[2][16][32][24];

  const int tid = threadIdx.x;
  const int lane = tid & 31;
  const int wave = tid >> 5;
  const int wm = wave & 1;   // 0..1 -> 64-row band
  const int wn = wave >> 1;  // 0..3 -> 64-col band
  const long long tile_m = (long long)blockIdx.x * 128;
  const long long tile_n = (long long)blockIdx.y * 256;

  const float scale = QDQ_OUT ? a_scales[sidx] : 1.0f;

  // ---- precompute staging coordinates (constant over the K loop) ----
  int a_row[2], a_k8[2], a_off[2];   // A: 512 chunks of 8 halves, 2/thread
  int b_row[4], b_k8[4], b_off[4];   // B: 1024 chunks, 4/thread
#pragma unroll
  for (int i = 0; i < 2; ++i) {
    int c = tid + i * 256;
    a_row[i] = c >> 2;
    a_k8[i] = (c & 3) * 8;
    int sub = a_row[i] >> 4;
    int l = (a_row[i] & 15) + (((a_k8[i] >> 3) & 1) << 4);
    int h0 = (a_k8[i] >> 4) << 3;
    a_off[i] = (sub * 32 + l) * 24 + h0;
  }
#pragma unroll
  for (int i = 0; i < 4; ++i) {
    int c = tid + i * 256;
    b_row[i] = c >> 2;
    b_k8[i] = (c & 3) * 8;
    int sub = b_row[i] >> 4;
    int l = (b_row[i] & 15) + (((b_k8[i] >> 3) & 1) << 4);
    int h0 = (b_k8[i] >> 4) << 3;
    b_off[i] = (sub * 32 + l) * 24 + h0;
  }

  v8f acc[4][4] = {};
  v8h ta[2], tb[4];

  // prime the pipeline: global loads for tile 0
#pragma unroll
  for (int i = 0; i < 2; ++i)
    ta[i] = *(const v8h*)(A + (tile_m + a_row[i]) * (long long)K + a_k8[i]);
#pragma unroll
  for (int i = 0; i < 4; ++i)
    tb[i] = *(const v8h*)(B + (tile_n + b_row[i]) * (long long)K + b_k8[i]);

  int buf = 0;
  for (int kt = 0; kt < K; kt += 32) {
    // stage current tile into LDS (fragment order)
#pragma unroll
    for (int i = 0; i < 2; ++i) *(v8h*)&lds_a[buf][0][0][a_off[i]] = ta[i];
#pragma unroll
    for (int i = 0; i < 4; ++i) *(v8h*)&lds_b[buf][0][0][b_off[i]] = tb[i];
    __syncthreads();

    // issue global loads for the next tile; they drain during the WMMAs
    if (kt + 32 < K) {
      const int kn = kt + 32;
#pragma unroll
      for (int i = 0; i < 2; ++i)
        ta[i] = *(const v8h*)(A + (tile_m + a_row[i]) * (long long)K + kn + a_k8[i]);
#pragma unroll
      for (int i = 0; i < 4; ++i)
        tb[i] = *(const v8h*)(B + (tile_n + b_row[i]) * (long long)K + kn + b_k8[i]);
      if (kn + 32 < K)  // deepen the weight stream into GL2
        __builtin_prefetch(B + (tile_n + (tid >> 1)) * (long long)K + kn + 32 +
                               (tid & 1) * 16, 0, 0);
    }

    // 16 WMMAs from the current buffer
    FragU bfrag[4];
#pragma unroll
    for (int j = 0; j < 4; ++j) {
      bfrag[j].h[0] = *(const v8h*)&lds_b[buf][wn * 4 + j][lane][0];
      bfrag[j].h[1] = *(const v8h*)&lds_b[buf][wn * 4 + j][lane][8];
    }
#pragma unroll
    for (int i = 0; i < 4; ++i) {
      FragU afrag;
      afrag.h[0] = *(const v8h*)&lds_a[buf][wm * 4 + i][lane][0];
      afrag.h[1] = *(const v8h*)&lds_a[buf][wm * 4 + i][lane][8];
#pragma unroll
      for (int j = 0; j < 4; ++j)
        acc[i][j] = __builtin_amdgcn_wmma_f32_16x16x32_f16(
            false, afrag.v, false, bfrag[j].v, (short)0, acc[i][j],
            false, false);
    }
    buf ^= 1;
  }

  // ---- epilogue: C layout -> m = r + 8*(lane>=16), n = lane&15 ----
  const int n_lo = lane & 15;
  const int m_hi = (lane >> 4) << 3;
#pragma unroll
  for (int i = 0; i < 4; ++i) {
#pragma unroll
    for (int j = 0; j < 4; ++j) {
      long long gn = tile_n + wn * 64 + j * 16 + n_lo;
#pragma unroll
      for (int r = 0; r < 8; ++r) {
        long long gm = tile_m + wm * 64 + i * 16 + m_hi + r;
        float v = acc[i][j][r];
        if (QDQ_OUT) {
          float q = rintf(v / scale) + 32768.0f;
          q = fminf(fmaxf(q, 0.0f), 65535.0f);
          ((_Float16*)Cout)[gm * N + gn] = (_Float16)((q - 32768.0f) * scale);
        } else {
          ((float*)Cout)[gm * N + gn] = v;
        }
      }
    }
  }
}

// ---------------------------------------------------------------------------
// Kernel 4: SiLU chain with sigmoid-QDQ (scale 1/65536, zp 0) + act QDQs.
// up/gate already QDQ'd (a1/a2) in the GEMM epilogues.
// ---------------------------------------------------------------------------
__global__ __launch_bounds__(256) void silu_o_kernel(
    const _Float16* __restrict__ up, const _Float16* __restrict__ gate,
    _Float16* __restrict__ o, const float* __restrict__ a_scales,
    long long n) {
  long long i = (blockIdx.x * 256LL + threadIdx.x) * 2;
  if (i >= n) return;
  float a3 = a_scales[3], a4 = a_scales[4];
#pragma unroll
  for (int j = 0; j < 2; ++j) {
    float g = (float)gate[i + j];
    float u = (float)up[i + j];
    float sg = 1.0f / (1.0f + expf(-g));
    float qs = fminf(fmaxf(rintf(sg * 65536.0f), 0.0f), 65535.0f) *
               (1.0f / 65536.0f);                      // sigmoid_qdq
    float act = g * qs;
    float q3 = fminf(fmaxf(rintf(act / a3) + 32768.0f, 0.0f), 65535.0f);
    act = (q3 - 32768.0f) * a3;                        // act_output_qdq
    float ov = act * u;
    float q4 = fminf(fmaxf(rintf(ov / a4) + 32768.0f, 0.0f), 65535.0f);
    o[i + j] = (_Float16)((q4 - 32768.0f) * a4);       // down_proj_input_qdq
  }
}

// ---------------------------------------------------------------------------
extern "C" void kernel_launch(void* const* d_in, const int* in_sizes, int n_in,
                              void* d_out, int out_size, void* d_ws,
                              size_t ws_size, hipStream_t stream) {
  const float* x       = (const float*)d_in[0];
  const float* w_gate  = (const float*)d_in[1];
  const float* s_gate  = (const float*)d_in[2];
  const float* w_up    = (const float*)d_in[3];
  const float* s_up    = (const float*)d_in[4];
  const float* w_down  = (const float*)d_in[5];
  const float* s_down  = (const float*)d_in[6];
  const float* a_sc    = (const float*)d_in[7];

  const long long S = in_sizes[0] / HID;   // 4096
  const long long nX  = S * HID;
  const long long nUG = S * INTER;

  // workspace layout (all sizes are multiples of 256B)
  char* ws = (char*)d_ws;
  size_t off = 0;
  _Float16* xq16 = (_Float16*)(ws + off); off += (size_t)nX * 2;
  _Float16* wu16 = (_Float16*)(ws + off); off += (size_t)INTER * HID * 2;
  _Float16* wg16 = (_Float16*)(ws + off); off += (size_t)INTER * HID * 2;
  _Float16* wd16 = (_Float16*)(ws + off); off += (size_t)HID * INTER * 2;
  _Float16* up16 = (_Float16*)(ws + off); off += (size_t)nUG * 2;
  _Float16* gt16 = (_Float16*)(ws + off); off += (size_t)nUG * 2;
  _Float16* o16  = (_Float16*)(ws + off); off += (size_t)nUG * 2;

  // 1) weight dequant (LPBQ fake-quant -> f16)
  {
    long long nb = (long long)INTER * HID / 16;
    int grid = (int)((nb + 255) / 256);
    lpbq_dequant_kernel<<<grid, 256, 0, stream>>>(w_up,   s_up,   wu16, HID,  nb);
    lpbq_dequant_kernel<<<grid, 256, 0, stream>>>(w_gate, s_gate, wg16, HID,  nb);
    long long nbd = (long long)HID * INTER / 16;
    int gridd = (int)((nbd + 255) / 256);
    lpbq_dequant_kernel<<<gridd, 256, 0, stream>>>(w_down, s_down, wd16, INTER, nbd);
  }
  // 2) x act-QDQ -> f16
  {
    int grid = (int)((nX / 4 + 255) / 256);
    act_qdq_kernel<<<grid, 256, 0, stream>>>(x, xq16, a_sc, nX);
  }
  // 3) up & gate projections with fused act-QDQ epilogue
  {
    dim3 grid((unsigned)(S / 128), (unsigned)(INTER / 256));
    gemm_wmma_kernel<true><<<grid, 256, 0, stream>>>(
        xq16, wu16, up16, (int)S, INTER, HID, a_sc, 1);
    gemm_wmma_kernel<true><<<grid, 256, 0, stream>>>(
        xq16, wg16, gt16, (int)S, INTER, HID, a_sc, 2);
  }
  // 4) SiLU + sigmoid-QDQ + act-QDQ chain -> o (f16)
  {
    int grid = (int)((nUG / 2 + 255) / 256);
    silu_o_kernel<<<grid, 256, 0, stream>>>(up16, gt16, o16, a_sc, nUG);
  }
  // 5) down projection -> f32 output
  {
    dim3 grid((unsigned)(S / 128), (unsigned)(HID / 256));
    gemm_wmma_kernel<false><<<grid, 256, 0, stream>>>(
        o16, wd16, (float*)d_out, (int)S, HID, INTER, a_sc, 0);
  }
}
